// FlashAttention2_53618371723497
// MI455X (gfx1250) — matrix operand order repaired and checked by reading, not compiled, not run on hardware
//
#include <hip/hip_runtime.h>
#include <cstdint>
#include <cstddef>

#define H_   16
#define KV_  4
#define D_   128
#define WIN_ 1024

typedef __attribute__((ext_vector_type(16))) __bf16 v16bf;
typedef __attribute__((ext_vector_type(8)))  float  v8f;

union BFrag {
  uint4 q[2];
  v16bf v;
};

__device__ __forceinline__ __bf16 f2bf(float x) {
  unsigned u = __float_as_uint(x);
  u += 0x7fffu + ((u >> 16) & 1u);            // round-to-nearest-even
  unsigned short h = (unsigned short)(u >> 16);
  return __builtin_bit_cast(__bf16, h);
}

__device__ __forceinline__ v8f wmma_bf16(v16bf a, v16bf b, v8f c) {
  return __builtin_amdgcn_wmma_f32_16x16x32_bf16(false, a, false, b, (short)0, c,
                                                 false, false);
}

// A-operand (16x32, M=lane&15): u0 = K[k0+8*hi .. +8), u1 = K[k0+16+8*hi .. +8)
__device__ __forceinline__ void load_a(BFrag& f, const __bf16* row, int k0, int hi) {
  const __bf16* p = row + k0 + hi * 8;
  f.q[0] = *reinterpret_cast<const uint4*>(p);
  f.q[1] = *reinterpret_cast<const uint4*>(p + 16);
}
// B-operand (32x16, N=lane&15): u0 = K[k0+8*hi .. +8), u1 = K[k0+16+8*hi .. +8)
__device__ __forceinline__ void load_b(BFrag& f, const __bf16* row, int k0, int hi) {
  const __bf16* p = row + k0 + hi * 8;
  f.q[0] = *reinterpret_cast<const uint4*>(p);
  f.q[1] = *reinterpret_cast<const uint4*>(p + 16);
}

// ---------------- elementwise conversion ----------------
__global__ __launch_bounds__(256) void krn_cvt_bf16(const float* __restrict__ x,
                                                    __bf16* __restrict__ y, int n) {
  int i = blockIdx.x * 256 + threadIdx.x;
  if (i < n) y[i] = f2bf(x[i]);
}

// ---------------- weight transpose + convert: W[k][n] -> WT[n][k] bf16 --------
__global__ __launch_bounds__(256) void krn_transpose_cvt(const float* __restrict__ w,
                                                         __bf16* __restrict__ wt,
                                                         int k, int n) {
  __shared__ float tile[32][33];
  int bn = blockIdx.x % (n >> 5);
  int bk = blockIdx.x / (n >> 5);
  int tx = threadIdx.x & 31;
  int ty = threadIdx.x >> 5;
  for (int i = ty; i < 32; i += 8)
    tile[i][tx] = w[(size_t)(bk * 32 + i) * n + bn * 32 + tx];
  __syncthreads();
  for (int i = ty; i < 32; i += 8)
    wt[(size_t)(bn * 32 + i) * k + bk * 32 + tx] = f2bf(tile[tx][i]);
}

// ---------------- RoPE + L2 norm + per-dim scale, f32 -> bf16 -----------------
// X rows: [(b*S+s)*NH + h][128]
__global__ __launch_bounds__(256) void krn_rope_norm(const float* __restrict__ x,
                                                     __bf16* __restrict__ y,
                                                     const float* __restrict__ scale,
                                                     int nh, int S, int nrows) {
  int lane = threadIdx.x & 31;
  int row = (blockIdx.x * 256 + threadIdx.x) >> 5;
  if (row >= nrows) return;
  int s = (row / nh) % S;
  const float* xp = x + (size_t)row * D_;
  float x0 = xp[lane], x1 = xp[lane + 32], x2 = xp[lane + 64], x3 = xp[lane + 96];
  const float LN1E4_O64 = 0.14391156830900384f;       // ln(10000)/64
  float f0 = __expf(-(float)lane * LN1E4_O64);
  float f1 = __expf(-(float)(lane + 32) * LN1E4_O64);
  float a0 = (float)s * f0, a1 = (float)s * f1;
  float c0 = __cosf(a0), sn0 = __sinf(a0);
  float c1 = __cosf(a1), sn1 = __sinf(a1);
  float r0 = x0 * c0 - x2 * sn0;
  float r1 = x1 * c1 - x3 * sn1;
  float r2 = x2 * c0 + x0 * sn0;
  float r3 = x3 * c1 + x1 * sn1;
  float ss = r0 * r0 + r1 * r1 + r2 * r2 + r3 * r3;
  for (int m = 16; m >= 1; m >>= 1) ss += __shfl_xor(ss, m, 32);
  float inv = 1.f / (sqrtf(ss) + 1e-6f);
  __bf16* yp = y + (size_t)row * D_;
  yp[lane]      = f2bf(r0 * inv * scale[lane]);
  yp[lane + 32] = f2bf(r1 * inv * scale[lane + 32]);
  yp[lane + 64] = f2bf(r2 * inv * scale[lane + 64]);
  yp[lane + 96] = f2bf(r3 * inv * scale[lane + 96]);
}

// ---------------- V: [B,S,KV,D] f32 -> [B,KV,D,S] bf16 -----------------------
__global__ __launch_bounds__(256) void krn_v_transpose(const float* __restrict__ vf,
                                                       __bf16* __restrict__ vt,
                                                       int B, int S) {
  int i = blockIdx.x * 256 + threadIdx.x;
  int total = B * S * KV_ * D_;
  if (i >= total) return;
  int d = i & (D_ - 1);
  int t = i >> 7;
  int kv = t & (KV_ - 1);
  t >>= 2;
  int s = t % S;
  int b = t / S;
  vt[(((size_t)(b * KV_ + kv)) * D_ + d) * S + s] = f2bf(vf[i]);
}

// ---------------- bf16 GEMM: C[M][N] = A[M][K] * BT[N][K]^T, f32 out ----------
// One wave per 32x32 tile (2x2 of 16x16), K-step 32.
__global__ __launch_bounds__(256) void krn_gemm_bf16(const __bf16* __restrict__ A,
                                                     const __bf16* __restrict__ BT,
                                                     float* __restrict__ C,
                                                     int M, int N, int K) {
  int lane = threadIdx.x & 31;
  int w = threadIdx.x >> 5;
  int tiles_n = N >> 5;
  int tile = blockIdx.x * 8 + w;
  if (tile >= (M >> 5) * tiles_n) return;
  int tm = tile / tiles_n, tn = tile - tm * tiles_n;
  int r = lane & 15, hi = lane >> 4;
  const __bf16* a0p = A + (size_t)(tm * 32 + r) * K;
  const __bf16* a1p = a0p + (size_t)16 * K;
  const __bf16* b0p = BT + (size_t)(tn * 32 + r) * K;
  const __bf16* b1p = b0p + (size_t)16 * K;
  v8f c00 = {0, 0, 0, 0, 0, 0, 0, 0};
  v8f c01 = c00, c10 = c00, c11 = c00;
  for (int k0 = 0; k0 < K; k0 += 32) {
    BFrag a0, a1, b0, b1;
    load_a(a0, a0p, k0, hi);
    load_a(a1, a1p, k0, hi);
    load_b(b0, b0p, k0, hi);
    load_b(b1, b1p, k0, hi);
    __builtin_prefetch(a0p + k0 + 256, 0, 0);
    __builtin_prefetch(b0p + k0 + 256, 0, 0);
    c00 = wmma_bf16(a0.v, b0.v, c00);
    c01 = wmma_bf16(a0.v, b1.v, c01);
    c10 = wmma_bf16(a1.v, b0.v, c10);
    c11 = wmma_bf16(a1.v, b1.v, c11);
  }
  size_t cb = (size_t)(tm * 32 + hi * 8) * N + tn * 32 + r;
  for (int i = 0; i < 8; i++) {
    C[cb + (size_t)i * N]            = c00[i];
    C[cb + (size_t)i * N + 16]       = c01[i];
    C[cb + (size_t)(i + 16) * N]      = c10[i];
    C[cb + (size_t)(i + 16) * N + 16] = c11[i];
  }
}

// ---------------- flash attention, sliding window + soft-cap ------------------
// One wave per (b, h, 16-query tile). Q:[B,S,H,D] K:[B,S,KV,D] Vt:[B,KV,D,S]
__global__ __launch_bounds__(256) void krn_attn(const __bf16* __restrict__ Qm,
                                                const __bf16* __restrict__ Km,
                                                const __bf16* __restrict__ Vt,
                                                __bf16* __restrict__ Om,
                                                int B, int S) {
  const float SCALE = 0.088388347648318447f;  // 1/sqrt(128)
  const float ICAP = 0.02f;                   // 1/50
  const float CAPF = 50.0f;
  const float NEG = -1e30f;
  int lane = threadIdx.x & 31;
  int w = threadIdx.x >> 5;
  int wg = blockIdx.x * 8 + w;
  int nqt = S >> 4;
  int qt = wg % nqt;
  int h = (wg / nqt) % H_;
  int b = wg / (nqt * H_);
  int q0 = qt << 4;
  int kvh = h >> 2;  // H/KV = 4
  int r = lane & 15, hi = lane >> 4;

  __shared__ __align__(16) __bf16 sh[8][16][32];  // P tile per wave

  // resident Q fragments (16 x 128 over four K=32 chunks)
  const __bf16* qrow = Qm + (((size_t)(b * S + q0 + r)) * H_ + h) * D_;
  BFrag qf[4];
  for (int c = 0; c < 4; c++) load_a(qf[c], qrow, c * 32, hi);

  v8f o[8];
  for (int c = 0; c < 8; c++) o[c] = v8f{0, 0, 0, 0, 0, 0, 0, 0};
  float mrun[8], lrun[8];
  for (int i = 0; i < 8; i++) { mrun[i] = NEG; lrun[i] = 0.f; }

  int jlo = q0 - WIN_; if (jlo < 0) jlo = 0; jlo &= ~31;
  int jhi = q0 + 16 + WIN_; jhi = (jhi + 31) & ~31; if (jhi > S) jhi = S;

  const __bf16* kbase = Km + ((size_t)(b * S) * KV_ + kvh) * D_;
  const __bf16* vbase = Vt + ((size_t)(b * KV_ + kvh)) * D_ * (size_t)S;

  for (int j0 = jlo; j0 < jhi; j0 += 32) {
    // ---- scores: two 16x16 tiles over keys j0..j0+31 ----
    v8f s0 = {0, 0, 0, 0, 0, 0, 0, 0};
    v8f s1 = s0;
    const __bf16* k0p = kbase + (size_t)(j0 + r) * (KV_ * D_);
    const __bf16* k1p = k0p + (size_t)16 * (KV_ * D_);
    for (int c = 0; c < 4; c++) {
      BFrag kb0, kb1;
      load_b(kb0, k0p, c * 32, hi);
      load_b(kb1, k1p, c * 32, hi);
      s0 = wmma_bf16(qf[c].v, kb0.v, s0);
      s1 = wmma_bf16(qf[c].v, kb1.v, s1);
    }
    // ---- soft-cap, window mask, online softmax ----
    float alpha[8];
    for (int i = 0; i < 8; i++) {
      int qi = q0 + i + hi * 8;
      float xa = s0[i] * SCALE; xa = CAPF * tanhf(xa * ICAP);
      float xb = s1[i] * SCALE; xb = CAPF * tanhf(xb * ICAP);
      int da = qi - (j0 + r);      if (da < 0) da = -da;
      int db = qi - (j0 + 16 + r); if (db < 0) db = -db;
      if (da > WIN_) xa = NEG;
      if (db > WIN_) xb = NEG;
      float mx = fmaxf(xa, xb);
      mx = fmaxf(mx, __shfl_xor(mx, 1, 32));
      mx = fmaxf(mx, __shfl_xor(mx, 2, 32));
      mx = fmaxf(mx, __shfl_xor(mx, 4, 32));
      mx = fmaxf(mx, __shfl_xor(mx, 8, 32));
      float mn = fmaxf(mrun[i], mx);
      float al = __expf(mrun[i] - mn);
      float pa = __expf(xa - mn);
      float pb = __expf(xb - mn);
      float rs = pa + pb;
      rs += __shfl_xor(rs, 1, 32);
      rs += __shfl_xor(rs, 2, 32);
      rs += __shfl_xor(rs, 4, 32);
      rs += __shfl_xor(rs, 8, 32);
      lrun[i] = lrun[i] * al + rs;
      mrun[i] = mn;
      alpha[i] = al;
      int m = i + hi * 8;
      sh[w][m][r] = f2bf(pa);        // C-layout -> LDS row-major
      sh[w][m][16 + r] = f2bf(pb);
    }
    for (int c = 0; c < 8; c++)
      for (int i = 0; i < 8; i++) o[c][i] *= alpha[i];
    // reload P as A-fragment (same-wave DS ops are in-order)
    BFrag pf;
    load_a(pf, &sh[w][r][0], 0, hi);
    // ---- O += P @ V ----
    for (int c = 0; c < 8; c++) {
      BFrag vfr;
      load_b(vfr, vbase + (size_t)(c * 16 + r) * S + j0, 0, hi);
      o[c] = wmma_bf16(pf.v, vfr.v, o[c]);
    }
  }
  // ---- finalize ----
  float inv[8];
  for (int i = 0; i < 8; i++) inv[i] = 1.f / lrun[i];
  for (int i = 0; i < 8; i++) {
    __bf16* op = Om + (((size_t)(b * S + q0 + i + hi * 8)) * H_ + h) * D_ + r;
    for (int c = 0; c < 8; c++) op[c * 16] = f2bf(o[c][i] * inv[i]);
  }
}

// =============================================================================
extern "C" void kernel_launch(void* const* d_in, const int* in_sizes, int n_in,
                              void* d_out, int out_size, void* d_ws, size_t ws_size,
                              hipStream_t stream) {
  (void)in_sizes; (void)n_in; (void)out_size; (void)ws_size;
  const int B = 2, S = 2048, E = 2048, H = 16, KV = 4, D = 128;
  const int M = B * S;

  const float* hs     = (const float*)d_in[0];
  const float* wq     = (const float*)d_in[1];
  const float* wk     = (const float*)d_in[2];
  const float* wv     = (const float*)d_in[3];
  const float* wo     = (const float*)d_in[4];
  const float* qscale = (const float*)d_in[5];
  const float* kscale = (const float*)d_in[6];
  float* out = (float*)d_out;

  char* wp = (char*)d_ws;
  auto alloc = [&](size_t bytes) -> void* {
    void* p = (void*)wp;
    wp += (bytes + 255) & ~(size_t)255;
    return p;
  };
  __bf16* hsb = (__bf16*)alloc((size_t)M * E * 2);
  __bf16* wqT = (__bf16*)alloc((size_t)E * H * D * 2);
  __bf16* wkT = (__bf16*)alloc((size_t)E * KV * D * 2);
  __bf16* wvT = (__bf16*)alloc((size_t)E * KV * D * 2);
  __bf16* woT = (__bf16*)alloc((size_t)H * D * E * 2);
  float*  Qf  = (float*)alloc((size_t)M * H * D * 4);
  float*  Kf  = (float*)alloc((size_t)M * KV * D * 4);
  float*  Vf  = (float*)alloc((size_t)M * KV * D * 4);
  __bf16* Qb  = (__bf16*)alloc((size_t)M * H * D * 2);
  __bf16* Kb  = (__bf16*)alloc((size_t)M * KV * D * 2);
  __bf16* Vt  = (__bf16*)alloc((size_t)M * KV * D * 2);
  __bf16* Ob  = (__bf16*)alloc((size_t)M * H * D * 2);

  // 1) hidden states -> bf16
  {
    int n = M * E;
    krn_cvt_bf16<<<(n + 255) / 256, 256, 0, stream>>>(hs, hsb, n);
  }
  // 2) weights -> transposed bf16 (B-operand layout)
  krn_transpose_cvt<<<(E / 32) * (H * D / 32), 256, 0, stream>>>(wq, wqT, E, H * D);
  krn_transpose_cvt<<<(E / 32) * (KV * D / 32), 256, 0, stream>>>(wk, wkT, E, KV * D);
  krn_transpose_cvt<<<(E / 32) * (KV * D / 32), 256, 0, stream>>>(wv, wvT, E, KV * D);
  krn_transpose_cvt<<<(H * D / 32) * (E / 32), 256, 0, stream>>>(wo, woT, H * D, E);
  // 3) Q/K/V projections (WMMA)
  krn_gemm_bf16<<<(M / 32) * (H * D / 32) / 8, 256, 0, stream>>>(hsb, wqT, Qf, M, H * D, E);
  krn_gemm_bf16<<<(M / 32) * (KV * D / 32) / 8, 256, 0, stream>>>(hsb, wkT, Kf, M, KV * D, E);
  krn_gemm_bf16<<<(M / 32) * (KV * D / 32) / 8, 256, 0, stream>>>(hsb, wvT, Vf, M, KV * D, E);
  // 4) RoPE + QK-norm epilogues
  krn_rope_norm<<<(M * H * 32) / 256, 256, 0, stream>>>(Qf, Qb, qscale, H, S, M * H);
  krn_rope_norm<<<(M * KV * 32) / 256, 256, 0, stream>>>(Kf, Kb, kscale, KV, S, M * KV);
  // 5) V -> transposed bf16 for PV B-fragments
  {
    int n = M * KV * D;
    krn_v_transpose<<<(n + 255) / 256, 256, 0, stream>>>(Vf, Vt, B, S);
  }
  // 6) sliding-window flash attention (WMMA QK^T and PV)
  krn_attn<<<(B * H * (S / 16)) / 8, 256, 0, stream>>>(Qb, Kb, Vt, Ob, B, S);
  // 7) output projection -> d_out (f32)
  krn_gemm_bf16<<<(M / 32) * (E / 32) / 8, 256, 0, stream>>>(Ob, woT, out, M, E, H * D);
}
